// CrossNetMix_65180423684674
// MI455X (gfx1250) — matrix operand order repaired and checked by reading, not compiled, not run on hardware
//
#include <hip/hip_runtime.h>

typedef __attribute__((ext_vector_type(16))) __bf16 v16bf;
typedef __attribute__((ext_vector_type(2)))  __bf16 v2bf;
typedef __attribute__((ext_vector_type(8)))  float  v8f;

// ---------------- ws layout (bf16 elements) ----------------
// VbC: [L][32][272][32]  GEMM1 B (+gating cols 256..271), per-k-chunk contiguous
// UbC: [L][64][16][256]  GEMM3 B, per-N-tile contiguous, column(d)-major over K=n
// Cb:  [L][4][64][64]    GEMM2 B, column(s)-major over K=r
#define OFF_VBC 0
#define OFF_UBC 835584        // 3*32*272*32
#define OFF_CB  1622016       // + 3*64*16*256
#define WS_TOTAL 1671168      // + 3*4*64*64

#define D_IN 1024
#define NEXP 4
#define RLOW 64
#define TSTRIDE 272           // T buffer row stride (bf16 elems)
#define ASTRIDE 40            // staged GEMM1-B column stride (32 + 8 pad)
#define USTRIDE 264           // staged GEMM3-B column stride (256 + 8 pad)

__device__ __forceinline__ unsigned short f2bf(float f) {
    unsigned int u = __float_as_uint(f);
    unsigned int r = u + 0x7FFFu + ((u >> 16) & 1u);   // RNE
    return (unsigned short)(r >> 16);
}

__device__ __forceinline__ unsigned int pk2bf(float a, float b) {
#if __has_builtin(__builtin_amdgcn_cvt_pk_bf16_f32)
    union { v2bf v; unsigned int u; } x;
    x.v = __builtin_amdgcn_cvt_pk_bf16_f32(a, b);
    return x.u;
#else
    return (unsigned int)f2bf(a) | ((unsigned int)f2bf(b) << 16);
#endif
}

__device__ __forceinline__ float fast_tanh(float x) {
#if __has_builtin(__builtin_amdgcn_tanhf)
    return __builtin_amdgcn_tanhf(x);
#else
    return tanhf(x);
#endif
}

union Frag {
    v16bf v;
    unsigned short h[16];
    unsigned int w[8];
    uint4 q[2];
};

__device__ __forceinline__ v8f wmma_bf16(const Frag& a, const Frag& b, v8f c) {
    return __builtin_amdgcn_wmma_f32_16x16x32_bf16(false, a.v, false, b.v,
                                                   (short)0, c, false, false);
}

// ---------------- weight repack: f32 -> bf16 staging-friendly layouts ----------------
__global__ void crossmix_prep(const float* __restrict__ U, const float* __restrict__ V,
                              const float* __restrict__ C, const float* __restrict__ G,
                              unsigned short* __restrict__ ws) {
    long idx = (long)blockIdx.x * blockDim.x + threadIdx.x;
    if (idx >= WS_TOTAL) return;
    float val;
    if (idx < OFF_UBC) {                      // VbC[l][c][n][k]
        long t = idx;
        int l = (int)(t / 278528); int rem = (int)(t % 278528);
        int c = rem / 8704; int rem2 = rem % 8704;
        int n = rem2 / 32, k = rem2 % 32;
        int d = c * 32 + k;
        if (n < 256) {
            int e = n >> 6, r = n & 63;
            val = V[(((long)(l * 4 + e) * 1024 + d) * 64) + r];
        } else {
            int e = n - 256;
            val = (e < 4) ? G[(long)e * 1024 + d] : 0.0f;
        }
    } else if (idx < OFF_CB) {                // UbC[l][nt][col][k]
        long t = idx - OFF_UBC;
        int l = (int)(t / 262144); int rem = (int)(t % 262144);
        int nt = rem / 4096; int rem2 = rem % 4096;
        int col = rem2 / 256, n = rem2 % 256;
        int d = nt * 16 + col;
        int e = n >> 6, r = n & 63;
        val = U[(((long)(l * 4 + e) * 1024 + d) * 64) + r];
    } else {                                  // Cb[l][e][s][r] = C[l,e,r,s]
        long t = idx - OFF_CB;
        int l = (int)(t / 16384); int rem = (int)(t % 16384);
        int e = rem / 4096; int rem2 = rem % 4096;
        int s = rem2 / 64, r = rem2 % 64;
        val = C[(((long)(l * 4 + e) * 64 + r) * 64) + s];
    }
    ws[idx] = f2bf(val);
}

// ---------------- one cross layer, fully fused ----------------
// block = 128 threads = 4 waves; each wave owns 16 rows; grid = B/64.
__global__ __launch_bounds__(128)
void crossmix_layer(const float* xi_in, const float* __restrict__ x0, float* out,
                    const unsigned short* __restrict__ VbC,
                    const unsigned short* __restrict__ UbC,
                    const unsigned short* __restrict__ Cb,
                    const float* __restrict__ bias) {
    __shared__ unsigned short Stage[272 * ASTRIDE];   // 21760 B (reused in phase C)
    __shared__ unsigned short Tlds[4][16 * TSTRIDE];  // 34816 B
    __shared__ float Glds[4][16 * 16];                //  4096 B
    __shared__ float Slds[4][16 * 4];                 //  1024 B

    const int tid  = threadIdx.x;
    const int w    = tid >> 5;
    const int lane = tid & 31;
    const int ln   = lane & 15;
    const int hi   = lane >> 4;
    const int rowBase = blockIdx.x * 64 + w * 16;
    const int row = rowBase + ln;

    unsigned short* Tw = &Tlds[w][0];

    // ---------------- Phase A: T = tanh(xi @ Vcat) + gating tile ----------------
    v8f acc[17];
    #pragma unroll
    for (int t = 0; t < 17; ++t) acc[t] = (v8f){0.f,0.f,0.f,0.f,0.f,0.f,0.f,0.f};

    // register double-buffer for cooperative B staging (272 cols x 32 k = 1088 x 16B)
    uint4 stg[9];
    {
        const unsigned short* src = VbC;
        #pragma unroll
        for (int i = 0; i < 9; ++i) {
            int id = tid + i * 128;
            if (id < 1088) stg[i] = *(const uint4*)(src + id * 8);
        }
    }

    const float* rowp = xi_in + (size_t)row * D_IN;
    for (int c = 0; c < 32; ++c) {
        // store staged chunk to LDS with column padding
        #pragma unroll
        for (int i = 0; i < 9; ++i) {
            int id = tid + i * 128;
            if (id < 1088) {
                int col = id >> 2, c4 = id & 3;
                *(uint4*)(&Stage[col * ASTRIDE + c4 * 8]) = stg[i];
            }
        }
        __syncthreads();
        // prefetch next chunk into registers while consuming this one
        if (c < 31) {
            const unsigned short* src = VbC + (size_t)(c + 1) * 8704;
            #pragma unroll
            for (int i = 0; i < 9; ++i) {
                int id = tid + i * 128;
                if (id < 1088) stg[i] = *(const uint4*)(src + id * 8);
            }
        }
        // A fragment: 16 rows of xi, f32 -> bf16
        const int k0 = c * 32 + hi * 16;
        const float* rp = rowp + k0;
        __builtin_prefetch(rp + 64, 0, 0);
        float4 f0 = *(const float4*)(rp);
        float4 f1 = *(const float4*)(rp + 4);
        float4 f2 = *(const float4*)(rp + 8);
        float4 f3 = *(const float4*)(rp + 12);
        Frag A;
        A.w[0] = pk2bf(f0.x, f0.y); A.w[1] = pk2bf(f0.z, f0.w);
        A.w[2] = pk2bf(f1.x, f1.y); A.w[3] = pk2bf(f1.z, f1.w);
        A.w[4] = pk2bf(f2.x, f2.y); A.w[5] = pk2bf(f2.z, f2.w);
        A.w[6] = pk2bf(f3.x, f3.y); A.w[7] = pk2bf(f3.z, f3.w);
        #pragma unroll
        for (int t = 0; t < 17; ++t) {
            const unsigned short* bp = &Stage[(t * 16 + ln) * ASTRIDE + hi * 16];
            Frag B;
            B.q[0] = *(const uint4*)(bp);
            B.q[1] = *(const uint4*)(bp + 8);
            acc[t] = wmma_bf16(A, B, acc[t]);
        }
        __syncthreads();
    }

    // tanh + bf16 store to LDS (C-layout -> row-major tile)
    #pragma unroll
    for (int t = 0; t < 16; ++t) {
        #pragma unroll
        for (int j = 0; j < 8; ++j) {
            int M = j + 8 * hi;
            Tw[M * TSTRIDE + t * 16 + ln] = f2bf(fast_tanh(acc[t][j]));
        }
    }
    #pragma unroll
    for (int j = 0; j < 8; ++j)
        Glds[w][(j + 8 * hi) * 16 + ln] = acc[16][j];

    // softmax over experts (lanes 0..15, one row each); wave-local, DS in order
    if (lane < 16) {
        float s0 = Glds[w][lane * 16 + 0];
        float s1 = Glds[w][lane * 16 + 1];
        float s2 = Glds[w][lane * 16 + 2];
        float s3 = Glds[w][lane * 16 + 3];
        float m = fmaxf(fmaxf(s0, s1), fmaxf(s2, s3));
        float e0 = __expf(s0 - m), e1 = __expf(s1 - m);
        float e2 = __expf(s2 - m), e3 = __expf(s3 - m);
        float inv = 1.0f / (e0 + e1 + e2 + e3);
        Slds[w][lane * 4 + 0] = e0 * inv;
        Slds[w][lane * 4 + 1] = e1 * inv;
        Slds[w][lane * 4 + 2] = e2 * inv;
        Slds[w][lane * 4 + 3] = e3 * inv;
    }

    // ---------------- Phase B: W' = score * tanh(T @ C)  (in place, per expert) ----------------
    #pragma unroll
    for (int e = 0; e < NEXP; ++e) {
        const unsigned short* tb = Tw + ln * TSTRIDE + e * RLOW;
        Frag A0, A1;
        A0.q[0] = *(const uint4*)(tb + hi * 16);
        A0.q[1] = *(const uint4*)(tb + hi * 16 + 8);
        A1.q[0] = *(const uint4*)(tb + 32 + hi * 16);
        A1.q[1] = *(const uint4*)(tb + 32 + hi * 16 + 8);

        v8f acc2[4];
        #pragma unroll
        for (int t = 0; t < 4; ++t) acc2[t] = (v8f){0.f,0.f,0.f,0.f,0.f,0.f,0.f,0.f};

        #pragma unroll
        for (int t = 0; t < 4; ++t) {
            const unsigned short* cb = Cb + e * 4096 + (t * 16 + ln) * 64;
            Frag B0, B1;
            B0.q[0] = *(const uint4*)(cb + hi * 16);
            B0.q[1] = *(const uint4*)(cb + hi * 16 + 8);
            acc2[t] = wmma_bf16(A0, B0, acc2[t]);
            B1.q[0] = *(const uint4*)(cb + 32 + hi * 16);
            B1.q[1] = *(const uint4*)(cb + 32 + hi * 16 + 8);
            acc2[t] = wmma_bf16(A1, B1, acc2[t]);
        }

        float sw[8];
        #pragma unroll
        for (int j = 0; j < 8; ++j) sw[j] = Slds[w][(j + 8 * hi) * 4 + e];

        #pragma unroll
        for (int t = 0; t < 4; ++t)
            #pragma unroll
            for (int j = 0; j < 8; ++j) {
                int M = j + 8 * hi;
                Tw[M * TSTRIDE + e * RLOW + t * 16 + ln] =
                    f2bf(fast_tanh(acc2[t][j]) * sw[j]);
            }
    }

    // ---------------- Phase C: Y = W' @ Ucat, staged B, fused epilogue ----------------
    Frag Wf[8];
    #pragma unroll
    for (int kk = 0; kk < 8; ++kk) {
        const unsigned short* tb = Tw + ln * TSTRIDE + kk * 32;
        Wf[kk].q[0] = *(const uint4*)(tb + hi * 16);
        Wf[kk].q[1] = *(const uint4*)(tb + hi * 16 + 8);
    }

    // stage one 16x256 Ub tile per nt (512 x 16B), register double-buffered
    uint4 stg2[4];
    {
        const unsigned short* src = UbC;
        #pragma unroll
        for (int i = 0; i < 4; ++i) stg2[i] = *(const uint4*)(src + (tid + i * 128) * 8);
    }

    for (int nt = 0; nt < 64; ++nt) {
        #pragma unroll
        for (int i = 0; i < 4; ++i) {
            int id = tid + i * 128;
            int col = id >> 5, cc = id & 31;
            *(uint4*)(&Stage[col * USTRIDE + cc * 8]) = stg2[i];
        }
        __syncthreads();
        if (nt < 63) {
            const unsigned short* src = UbC + (size_t)(nt + 1) * 4096;
            #pragma unroll
            for (int i = 0; i < 4; ++i) stg2[i] = *(const uint4*)(src + (tid + i * 128) * 8);
        }
        v8f acc3 = (v8f){0.f,0.f,0.f,0.f,0.f,0.f,0.f,0.f};
        #pragma unroll
        for (int kk = 0; kk < 8; ++kk) {
            const unsigned short* ub = &Stage[ln * USTRIDE + kk * 32 + hi * 16];
            Frag B;
            B.q[0] = *(const uint4*)(ub);
            B.q[1] = *(const uint4*)(ub + 8);
            acc3 = wmma_bf16(Wf[kk], B, acc3);
        }
        const int d = nt * 16 + ln;
        const float bv = bias[d];
        #pragma unroll
        for (int j = 0; j < 8; ++j) {
            const int M = j + 8 * hi;
            const size_t idx = (size_t)(rowBase + M) * D_IN + d;
            out[idx] = x0[idx] * (acc3[j] + bv) + xi_in[idx];
        }
        __syncthreads();
    }
}

extern "C" void kernel_launch(void* const* d_in, const int* in_sizes, int n_in,
                              void* d_out, int out_size, void* d_ws, size_t ws_size,
                              hipStream_t stream) {
    (void)in_sizes; (void)n_in; (void)out_size; (void)ws_size;
    const float* x      = (const float*)d_in[0];
    const float* U      = (const float*)d_in[1];
    const float* V      = (const float*)d_in[2];
    const float* C      = (const float*)d_in[3];
    const float* biases = (const float*)d_in[4];
    const float* G      = (const float*)d_in[5];
    float* out = (float*)d_out;
    unsigned short* ws = (unsigned short*)d_ws;

    // repack weights to bf16 staging layouts (~3.3 MB, L2-resident)
    crossmix_prep<<<WS_TOTAL / 256, 256, 0, stream>>>(U, V, C, G, ws);

    // 3 cross layers; d_out doubles as the xi buffer (layer 0 reads x)
    for (int l = 0; l < 3; ++l) {
        const float* xi = (l == 0) ? x : (const float*)out;
        crossmix_layer<<<32768 / 64, 128, 0, stream>>>(
            xi, x, out,
            ws + OFF_VBC + (size_t)l * 32 * 272 * 32,
            ws + OFF_UBC + (size_t)l * 64 * 16 * 256,
            ws + OFF_CB  + (size_t)l * 4 * 64 * 64,
            biases + (size_t)l * 1024);
    }
}
